// SelfAttention_47631187313038
// MI455X (gfx1250) — matrix operand order repaired and checked by reading, not compiled, not run on hardware
//
#include <hip/hip_runtime.h>
#include <hip/hip_bf16.h>

// ---------------------------------------------------------------------------
// Causal linear attention for MI455X / gfx1250.
//  - all matmuls on v_wmma_f32_16x16x32_f16 (f16 in, f32 accumulate)
//  - every WMMA operand read is lane-contiguous ds_load_b128
//  - pure-copy LDS staging uses GLOBAL_LOAD_ASYNC_TO_LDS_B128 (ASYNCcnt),
//    double-buffered so the async DMA overlaps the WMMAs
// ---------------------------------------------------------------------------

typedef __attribute__((ext_vector_type(16))) _Float16 v16h;
typedef __attribute__((ext_vector_type(8)))  _Float16 v8h;
typedef __attribute__((ext_vector_type(8)))  float    v8f;
typedef __attribute__((ext_vector_type(4)))  float    v4f;

#define WMMA_F32_F16(a, b, c) \
  __builtin_amdgcn_wmma_f32_16x16x32_f16(false, (a), false, (b), (short)0, (c), false, false)

constexpr int kB   = 2;
constexpr int kT   = 2048;
constexpr int kDIM = 512;
constexpr int kH   = 8;
constexpr int kD   = 64;
constexpr int kM   = kB * kT;
constexpr float kEPS = 1e-6f;

// ---- CDNA5 async global->LDS copy (16B per lane), tracked by ASYNCcnt ------
__device__ __forceinline__ void async_copy_b128(void* lds_dst, const void* g_src) {
  // LDS aperture: low 32 bits of the generic pointer are the LDS byte address.
  const uint32_t l = (uint32_t)(uintptr_t)lds_dst;
  const uint64_t g = (uint64_t)(uintptr_t)g_src;
  asm volatile("global_load_async_to_lds_b128 %0, %1, off" :: "v"(l), "v"(g) : "memory");
}
__device__ __forceinline__ void wait_async0() {
  asm volatile("s_wait_asynccnt 0" ::: "memory");
}

__device__ __forceinline__ v16h mk16(v8h lo, v8h hi) {
  return __builtin_shufflevector(lo, hi, 0, 1, 2, 3, 4, 5, 6, 7, 8, 9, 10, 11, 12, 13, 14, 15);
}

// ---------------- WMMA operand loaders (all lane-contiguous) ----------------

// A (16x32 f16) from row-major buf
__device__ __forceinline__ v16h load_a16(const _Float16* buf, int ld, int row0, int k0, int lane) {
  const int m  = row0 + (lane & 15);
  const int hb = (lane >> 4) * 16;
  const _Float16* p = buf + m * ld + k0 + hb;
  return mk16(*(const v8h*)p, *(const v8h*)(p + 8));
}

// B (32x16 f16) where B[k][n] = buf[n*ld + k]  (buf is N-major / transposed)
__device__ __forceinline__ v16h load_bT16(const _Float16* buf, int ld, int k0, int n0, int lane) {
  const int n  = n0 + (lane & 15);
  const int kh = (lane >> 4) * 16;
  const _Float16* p = buf + n * ld + k0 + kh;
  return mk16(*(const v8h*)p, *(const v8h*)(p + 8));
}

// B (32x16) from an N-major f32 buffer (cvt on load; used for the state S)
__device__ __forceinline__ v16h load_bT_f32(const float* buf, int ld, int k0, int n0, int lane) {
  const int n  = n0 + (lane & 15);
  const int kh = (lane >> 4) * 16;
  const float* p = buf + n * ld + k0 + kh;
  v4f f0 = *(const v4f*)p, f1 = *(const v4f*)(p + 4);
  v4f f2 = *(const v4f*)(p + 8), f3 = *(const v4f*)(p + 12);
  v16h b;
#pragma unroll
  for (int j = 0; j < 4; ++j) {
    b[j]      = (_Float16)f0[j];
    b[4 + j]  = (_Float16)f1[j];
    b[8 + j]  = (_Float16)f2[j];
    b[12 + j] = (_Float16)f3[j];
  }
  return b;
}

// C/D (16x16 f32) vs N-major f32 buffer: per-lane contiguous 8 floats
__device__ __forceinline__ v8f load_cT_f32(const float* buf, int ld, int row0, int col0, int lane) {
  const int n  = col0 + (lane & 15);
  const int rb = row0 + (lane >> 4) * 8;
  const float* p = buf + n * ld + rb;
  v4f c0 = *(const v4f*)p, c1 = *(const v4f*)(p + 4);
  return __builtin_shufflevector(c0, c1, 0, 1, 2, 3, 4, 5, 6, 7);
}

__device__ __forceinline__ void store_dT_f32(float* buf, int ld, int row0, int col0, v8f d, int lane) {
  const int n  = col0 + (lane & 15);
  const int rb = row0 + (lane >> 4) * 8;
  float* p = buf + n * ld + rb;
  *(v4f*)p       = __builtin_shufflevector(d, d, 0, 1, 2, 3);
  *(v4f*)(p + 4) = __builtin_shufflevector(d, d, 4, 5, 6, 7);
}

// D store to row-major f32 (strided; only for the small Cs epilogue tile)
__device__ __forceinline__ void store_d_f32(float* buf, int ld, int row0, int col0, v8f d, int lane) {
  const int n  = col0 + (lane & 15);
  const int rb = row0 + (lane >> 4) * 8;
#pragma unroll
  for (int r = 0; r < 8; ++r) buf[(rb + r) * ld + n] = d[r];
}

// ---------------- kernel 1: weight f32 -> f16 (4 matrices packed) ----------

__global__ __launch_bounds__(256) void cvt_w_kernel(const float* __restrict__ w0,
                                                    const float* __restrict__ w1,
                                                    const float* __restrict__ w2,
                                                    const float* __restrict__ w3,
                                                    _Float16* __restrict__ o) {
  const int w = blockIdx.y;
  const float* s = (w == 0) ? w0 : (w == 1) ? w1 : (w == 2) ? w2 : w3;
  const int i = (blockIdx.x * 256 + threadIdx.x) * 8;
  v4f a = *(const v4f*)(s + i), b = *(const v4f*)(s + i + 4);
  v8h h;
#pragma unroll
  for (int j = 0; j < 4; ++j) { h[j] = (_Float16)a[j]; h[4 + j] = (_Float16)b[j]; }
  *(v8h*)(o + (size_t)w * (kDIM * kDIM) + i) = h;
}

// ---------------- kernel 2: QKV projection GEMM + activation epilogue ------
// grid = (M/64, DIM/64, 3); N-tile == one head so softmax stays local.
// W panel: async double-buffered copy. x panel: VGPR path (needs f32->f16).

__global__ __launch_bounds__(256) void proj_kernel(const float* __restrict__ x,
                                                   const _Float16* __restrict__ w16,
                                                   _Float16* __restrict__ qkv) {
  __shared__ alignas(16) _Float16 As[2][64 * 32];
  __shared__ alignas(16) _Float16 Ws[2][64 * 32];
  __shared__ alignas(16) float    Cs[64 * 64];

  const int tid = threadIdx.x, lane = tid & 31, wave = tid >> 5;
  const int m0 = blockIdx.x * 64, n0 = blockIdx.y * 64, z = blockIdx.z;
  const _Float16* W = w16 + (size_t)z * (kDIM * kDIM);
  _Float16* outp = qkv + (size_t)z * ((size_t)kM * kDIM);
  const int mt = wave & 3, nh = wave >> 2;
  const int srow = tid >> 2;        // 0..63
  const int scol = (tid & 3) * 8;   // 0,8,16,24
  const int soff = srow * 32 + scol;
  const float*    xrow = x + (size_t)(m0 + srow) * kDIM + scol;
  const _Float16* wrow = W + (size_t)(n0 + srow) * kDIM + scol;

  auto stage_a = [&](int buf, int k0) {
    v4f x0 = *(const v4f*)(xrow + k0), x1 = *(const v4f*)(xrow + k0 + 4);
    v8h h;
#pragma unroll
    for (int j = 0; j < 4; ++j) { h[j] = (_Float16)x0[j]; h[4 + j] = (_Float16)x1[j]; }
    *(v8h*)(As[buf] + soff) = h;
  };

  // prologue: stage buffer 0
  async_copy_b128(Ws[0] + soff, wrow);
  stage_a(0, 0);
  wait_async0();
  __syncthreads();

  v8f acc[2] = {};
  for (int kt = 0; kt < kDIM / 32; ++kt) {
    const int cur = kt & 1;
    if (kt + 1 < kDIM / 32) {
      const int k0n = (kt + 1) * 32;
      async_copy_b128(Ws[cur ^ 1] + soff, wrow + k0n);
      stage_a(cur ^ 1, k0n);
      __builtin_prefetch(xrow + k0n + 32, 0, 1);
    }
    v16h a = load_a16(As[cur], 32, mt * 16, 0, lane);
#pragma unroll
    for (int i = 0; i < 2; ++i) {
      v16h b = load_bT16(Ws[cur], 32, 0, (nh * 2 + i) * 16, lane);
      acc[i] = WMMA_F32_F16(a, b, acc[i]);
    }
    wait_async0();
    __syncthreads();
  }
#pragma unroll
  for (int i = 0; i < 2; ++i)
    store_d_f32(Cs, 64, mt * 16, (nh * 2 + i) * 16, acc[i], lane);
  __syncthreads();

  if (z == 0) {
    if (tid < 64) {   // softmax over the head's 64 columns, one thread per row
      float mx = -1e30f;
      for (int c = 0; c < 64; ++c) mx = fmaxf(mx, Cs[tid * 64 + c]);
      float s = 0.f;
      for (int c = 0; c < 64; ++c) s += __expf(Cs[tid * 64 + c] - mx);
      const float inv = 1.f / s;
      for (int c = 0; c < 64; c += 8) {
        v8h h;
#pragma unroll
        for (int j = 0; j < 8; ++j) h[j] = (_Float16)(__expf(Cs[tid * 64 + c + j] - mx) * inv);
        *(v8h*)(outp + (size_t)(m0 + tid) * kDIM + n0 + c) = h;
      }
    }
  } else {
    const int r = tid >> 2, c0 = (tid & 3) * 16;
#pragma unroll
    for (int c = 0; c < 16; c += 8) {
      v8h h;
#pragma unroll
      for (int j = 0; j < 8; ++j) {
        float v = Cs[r * 64 + c0 + c + j];
        if (z == 1) v = (v > 0.f) ? (v + 1.f) : __expf(v);   // elu(k)+1
        h[j] = (_Float16)v;
      }
      *(v8h*)(outp + (size_t)(m0 + r) * kDIM + n0 + c0 + c) = h;
    }
  }
}

// ---------------- kernel 3: chunked causal linear-attention scan -----------
// One WG per (b,h), chunk C=64. qb/kb tiles arrive via async-to-LDS DMA while
// the kbT/vbT transposes go through VGPRs. All WMMA operand reads contiguous.

__global__ __launch_bounds__(256) void scan_kernel(const _Float16* __restrict__ q16,
                                                   const _Float16* __restrict__ k16,
                                                   const _Float16* __restrict__ v16,
                                                   _Float16* __restrict__ attn16) {
  __shared__ alignas(16) float    SshT[64 * 64];  // state, e-major: S[d][e] at [e*64+d]
  __shared__ alignas(16) float    zrun[64];
  __shared__ alignas(16) _Float16 qb [64 * 64];   // q'  [t][d]
  __shared__ alignas(16) _Float16 kb [64 * 64];   // k   [t][d]
  __shared__ alignas(16) _Float16 kbT[64 * 64];   // k   [d][t]
  __shared__ alignas(16) _Float16 vbT[64 * 64];   // v   [e][t]
  __shared__ alignas(16) _Float16 ab [64 * 64];   // scores / attn staging [t][*]

  const int tid = threadIdx.x, lane = tid & 31, wave = tid >> 5;
  const int bh = blockIdx.x, bb = bh / kH, hh = bh % kH;
  const size_t rowbase = (size_t)bb * kT;
  const int col0 = hh * kD;
  const int mt = wave & 3, nh = wave >> 2;
  const int tr = tid >> 2;          // chunk position handled by this thread
  const int d0 = (tid & 3) * 16;

#pragma unroll
  for (int j = 0; j < 16; ++j) SshT[tid + j * 256] = 0.f;
  if (tid < 64) zrun[tid] = 0.f;
  __syncthreads();

  for (int ch = 0; ch < kT / 64; ++ch) {
    const int t0 = ch * 64;
    // ---- chunk load: qb/kb via async DMA; k,v through VGPRs for transposes ----
    {
      const size_t src = (rowbase + t0 + tr) * kDIM + col0 + d0;
      async_copy_b128(qb + tr * 64 + d0,     q16 + src);
      async_copy_b128(qb + tr * 64 + d0 + 8, q16 + src + 8);
      async_copy_b128(kb + tr * 64 + d0,     k16 + src);
      async_copy_b128(kb + tr * 64 + d0 + 8, k16 + src + 8);
      v8h b0 = *(const v8h*)(k16 + src), b1 = *(const v8h*)(k16 + src + 8);
#pragma unroll
      for (int j = 0; j < 8; ++j) {
        kbT[(d0 + j) * 64 + tr] = b0[j];
        kbT[(d0 + 8 + j) * 64 + tr] = b1[j];
      }
      v8h c0 = *(const v8h*)(v16 + src), c1 = *(const v8h*)(v16 + src + 8);
#pragma unroll
      for (int j = 0; j < 8; ++j) {
        vbT[(d0 + j) * 64 + tr] = c0[j];
        vbT[(d0 + 8 + j) * 64 + tr] = c1[j];
      }
      wait_async0();
    }
    __syncthreads();
    // ---- inclusive z-cumsum (contiguous via kbT) + fold divide into q' ----
    if (tid < 64) {
      const int d = tid;
      float acc = zrun[d];
      for (int t = 0; t < 64; ++t) {
        acc += (float)kbT[d * 64 + t];
        const float qv = (float)qb[t * 64 + d];
        qb[t * 64 + d] = (_Float16)(qv / (acc + kEPS));
      }
      zrun[d] = acc;
    }
    __syncthreads();
    // ---- A = tril_incl(Q' K^T) ----
#pragma unroll
    for (int i = 0; i < 2; ++i) {
      const int nt = nh * 2 + i;
      v8f acc = {};
#pragma unroll
      for (int ks = 0; ks < 64; ks += 32) {
        v16h a = load_a16(qb, 64, mt * 16, ks, lane);
        v16h b = load_bT16(kb, 64, ks, nt * 16, lane);   // B[d][s] = kb[s][d]
        acc = WMMA_F32_F16(a, b, acc);
      }
      const int n = lane & 15, rb = (lane >> 4) * 8;
#pragma unroll
      for (int r = 0; r < 8; ++r) {
        const int rowi = mt * 16 + rb + r, coli = nt * 16 + n;
        ab[rowi * 64 + coli] = (_Float16)((coli <= rowi) ? acc[r] : 0.f);
      }
    }
    __syncthreads();
    // ---- attn = A V + Q' S_prev  and  dS = K^T V  (registers only) ----
    v8f accA[2], accS[2];
#pragma unroll
    for (int i = 0; i < 2; ++i) {
      const int nt = nh * 2 + i;
      v8f acc = {};
#pragma unroll
      for (int ks = 0; ks < 64; ks += 32) {
        v16h a = load_a16(ab, 64, mt * 16, ks, lane);
        v16h b = load_bT16(vbT, 64, ks, nt * 16, lane);  // B[s][e] = vbT[e][s]
        acc = WMMA_F32_F16(a, b, acc);
      }
#pragma unroll
      for (int ks = 0; ks < 64; ks += 32) {
        v16h a = load_a16(qb, 64, mt * 16, ks, lane);
        v16h b = load_bT_f32(SshT, 64, ks, nt * 16, lane);  // B[d][e] = SshT[e][d]
        acc = WMMA_F32_F16(a, b, acc);
      }
      accA[i] = acc;
      v8f s = load_cT_f32(SshT, 64, mt * 16, nt * 16, lane);
#pragma unroll
      for (int ks = 0; ks < 64; ks += 32) {
        v16h a = load_a16(kbT, 64, mt * 16, ks, lane);   // A[d][t] = kbT[d][t]
        v16h b = load_bT16(vbT, 64, ks, nt * 16, lane);
        s = WMMA_F32_F16(a, b, s);
      }
      accS[i] = s;
    }
    __syncthreads();
    // ---- commit: S update (contiguous f32) + attn tile into LDS staging ----
#pragma unroll
    for (int i = 0; i < 2; ++i) {
      const int nt = nh * 2 + i;
      store_dT_f32(SshT, 64, mt * 16, nt * 16, accS[i], lane);
      const int n = lane & 15, rb = (lane >> 4) * 8;
#pragma unroll
      for (int r = 0; r < 8; ++r)
        ab[(mt * 16 + rb + r) * 64 + nt * 16 + n] = (_Float16)accA[i][r];
    }
    __syncthreads();
    // ---- coalesced vectorized global store of the attn chunk ----
    {
      const size_t dst = (rowbase + t0 + tr) * kDIM + col0 + d0;
      *(v8h*)(attn16 + dst)     = *(const v8h*)(ab + tr * 64 + d0);
      *(v8h*)(attn16 + dst + 8) = *(const v8h*)(ab + tr * 64 + d0 + 8);
    }
  }
}

// ---------------- kernel 4: output GEMM + bias -----------------------------
// Both panels are pure f16 copies -> fully async double-buffered staging.

__global__ __launch_bounds__(256) void out_kernel(const _Float16* __restrict__ attn16,
                                                  const _Float16* __restrict__ wo16,
                                                  const float* __restrict__ bo,
                                                  float* __restrict__ out) {
  __shared__ alignas(16) _Float16 As[2][64 * 32];
  __shared__ alignas(16) _Float16 Ws[2][64 * 32];
  __shared__ alignas(16) float    Cs[64 * 64];

  const int tid = threadIdx.x, lane = tid & 31, wave = tid >> 5;
  const int m0 = blockIdx.x * 64, n0 = blockIdx.y * 64;
  const int mt = wave & 3, nh = wave >> 2;
  const int srow = tid >> 2, scol = (tid & 3) * 8;
  const int soff = srow * 32 + scol;
  const _Float16* arow = attn16 + (size_t)(m0 + srow) * kDIM + scol;
  const _Float16* wrow = wo16 + (size_t)(n0 + srow) * kDIM + scol;

  // prologue
  async_copy_b128(As[0] + soff, arow);
  async_copy_b128(Ws[0] + soff, wrow);
  wait_async0();
  __syncthreads();

  v8f acc[2] = {};
  for (int kt = 0; kt < kDIM / 32; ++kt) {
    const int cur = kt & 1;
    if (kt + 1 < kDIM / 32) {
      const int k0n = (kt + 1) * 32;
      async_copy_b128(As[cur ^ 1] + soff, arow + k0n);
      async_copy_b128(Ws[cur ^ 1] + soff, wrow + k0n);
    }
    v16h a = load_a16(As[cur], 32, mt * 16, 0, lane);
#pragma unroll
    for (int i = 0; i < 2; ++i) {
      v16h b = load_bT16(Ws[cur], 32, 0, (nh * 2 + i) * 16, lane);
      acc[i] = WMMA_F32_F16(a, b, acc[i]);
    }
    wait_async0();
    __syncthreads();
  }
#pragma unroll
  for (int i = 0; i < 2; ++i)
    store_d_f32(Cs, 64, mt * 16, (nh * 2 + i) * 16, acc[i], lane);
  __syncthreads();
  const int r = tid >> 2, c0 = (tid & 3) * 16;
#pragma unroll
  for (int c = 0; c < 16; c += 4) {
    v4f vv = *(const v4f*)(Cs + r * 64 + c0 + c);
    v4f bb = *(const v4f*)(bo + n0 + c0 + c);
    vv += bb;
    *(v4f*)(out + (size_t)(m0 + r) * kDIM + n0 + c0 + c) = vv;
  }
}

// ---------------- host launcher --------------------------------------------

extern "C" void kernel_launch(void* const* d_in, const int* in_sizes, int n_in,
                              void* d_out, int out_size, void* d_ws, size_t ws_size,
                              hipStream_t stream) {
  (void)in_sizes; (void)n_in; (void)out_size; (void)ws_size;
  const float* x  = (const float*)d_in[0];
  const float* Wq = (const float*)d_in[1];
  const float* Wk = (const float*)d_in[2];
  const float* Wv = (const float*)d_in[3];
  const float* Wo = (const float*)d_in[4];
  const float* bo = (const float*)d_in[5];
  float* out = (float*)d_out;

  // workspace (f16): [Wq|Wk|Wv|Wo] 2 MB | q 4 MB | k | v | attn  ~= 19 MB
  char* ws = (char*)d_ws;
  _Float16* w16 = (_Float16*)ws;
  _Float16* q16 = (_Float16*)(ws + (size_t)4 * kDIM * kDIM * sizeof(_Float16));
  _Float16* k16 = q16 + (size_t)kM * kDIM;
  _Float16* v16 = k16 + (size_t)kM * kDIM;
  _Float16* a16 = v16 + (size_t)kM * kDIM;

  cvt_w_kernel<<<dim3((kDIM * kDIM) / 2048, 4, 1), 256, 0, stream>>>(Wq, Wk, Wv, Wo, w16);
  proj_kernel<<<dim3(kM / 64, kDIM / 64, 3), 256, 0, stream>>>(x, w16, q16);
  scan_kernel<<<dim3(kB * kH, 1, 1), 256, 0, stream>>>(q16, k16, v16, a16);
  out_kernel<<<dim3(kM / 64, kDIM / 64, 1), 256, 0, stream>>>(
      a16, w16 + (size_t)3 * kDIM * kDIM, bo, out);
}
